// AudioTokenPredictor_50672024158657
// MI455X (gfx1250) — compile-verified, matrix operand order checked by reading, not run on hardware
//
#include <hip/hip_runtime.h>

// ---------------------------------------------------------------------------
// AudioTokenPredictor forward for MI455X (gfx1250, wave32, WMMA + TDM).
// GEMMs:  C[M,N] = A_bf16[M,K] * W_fp32[N,K]^T  via v_wmma_f32_16x16x32_bf16,
//         double-buffered LDS: A tiles DMA'd by the Tensor Data Mover while
//         the VALU converts the next W tile fp32->bf16 and the matrix pipe
//         computes the current tile (one barrier per K-tile).
// Attention: flash-style online softmax (DPP-butterfly reductions on the
//         VALU), QK^T and P*V via WMMA, K tiles via TDM, double-buffered.
// ---------------------------------------------------------------------------

constexpr int kL  = 12;
constexpr int kH  = 12;
constexpr int kD  = 768;
constexpr int kDH = 64;
constexpr int kFF = 3072;
constexpr int kS  = 8;
constexpr int kV  = 1024;
constexpr int kB  = 2;
constexpr int kT  = 1024;
constexpr int kM  = kB * kT;   // 2048 activation rows

typedef __attribute__((ext_vector_type(16))) __bf16 bf16x16;
typedef __attribute__((ext_vector_type(8)))  __bf16 bf16x8;
typedef __attribute__((ext_vector_type(8)))  float  f32x8;
typedef __attribute__((ext_vector_type(4)))  unsigned int u32x4;
typedef __attribute__((ext_vector_type(8)))  int    i32x8;
typedef __attribute__((ext_vector_type(4)))  int    i32x4;

__device__ __forceinline__ f32x8 wmma_bf16(bf16x16 a, bf16x16 b, f32x8 c) {
  // (neg_a, A, neg_b, B, c_mod, C, reuse_a, reuse_b)
  return __builtin_amdgcn_wmma_f32_16x16x32_bf16(false, a, false, b, (short)0, c,
                                                 false, false);
}

// Build a 16-half WMMA operand fragment from two contiguous 8-half groups.
// 16-bit operand layout (ISA 7.12.2): lane l (row/col = l%16) holds
// K = {base..base+7} in halves [0..7] and K = {base+16..base+23} in [8..15],
// base = (l<16) ? 0 : 8.  Same gather for A rows and NT-B (W rows).
__device__ __forceinline__ bf16x16 ldfrag(const __bf16* p0, const __bf16* p1) {
  bf16x8 lo = *(const bf16x8*)p0;
  bf16x8 hi = *(const bf16x8*)p1;
  bf16x16 r;
#pragma unroll
  for (int e = 0; e < 8; ++e) { r[e] = lo[e]; r[e + 8] = hi[e]; }
  return r;
}

// ---------------------------------------------------------------------------
// VALU-only 16-lane butterfly reductions via DPP (keeps the LDS pipe free for
// WMMA fragment loads).  Masks {1,2} = quad_perm, {4} = DPP8, {15} =
// row_mirror; span{1,2,4,15} = all 16 lanes, bit 4 untouched so the two
// 16-lane halves (C-matrix rows m and m+8) reduce independently.
// ---------------------------------------------------------------------------
#define kDpp8Xor4 6852524  // lanes [4,5,6,7,0,1,2,3] packed 3b each
__device__ __forceinline__ float red16_max(float x) {
  int i;
  i = __builtin_amdgcn_update_dpp(__float_as_int(x), __float_as_int(x),
                                  0xB1, 0xF, 0xF, false);        // quad xor1
  x = fmaxf(x, __int_as_float(i));
  i = __builtin_amdgcn_update_dpp(__float_as_int(x), __float_as_int(x),
                                  0x4E, 0xF, 0xF, false);        // quad xor2
  x = fmaxf(x, __int_as_float(i));
  i = __builtin_amdgcn_mov_dpp8(__float_as_int(x), kDpp8Xor4);   // xor4 in 8
  x = fmaxf(x, __int_as_float(i));
  i = __builtin_amdgcn_update_dpp(__float_as_int(x), __float_as_int(x),
                                  0x140, 0xF, 0xF, false);       // row_mirror
  x = fmaxf(x, __int_as_float(i));
  return x;
}
__device__ __forceinline__ float red16_sum(float x) {
  int i;
  i = __builtin_amdgcn_update_dpp(__float_as_int(x), __float_as_int(x),
                                  0xB1, 0xF, 0xF, false);
  x += __int_as_float(i);
  i = __builtin_amdgcn_update_dpp(__float_as_int(x), __float_as_int(x),
                                  0x4E, 0xF, 0xF, false);
  x += __int_as_float(i);
  i = __builtin_amdgcn_mov_dpp8(__float_as_int(x), kDpp8Xor4);
  x += __int_as_float(i);
  i = __builtin_amdgcn_update_dpp(__float_as_int(x), __float_as_int(x),
                                  0x140, 0xF, 0xF, false);
  x += __int_as_float(i);
  return x;
}

// ---------------------------------------------------------------------------
// Tensor Data Mover: DMA a 2D tile of 2-byte elements (tile_d0 elems per row,
// tile_d1 rows, row stride stride_elems) from global memory into LDS.
// D# per ISA 8.3-8.6: group0 = {count=1, lds_addr, global_addr, type=2},
// group1 = {data_size=2B, huge tensor dims (no OOB clip), tile dims, stride}.
// Issue from ONE wave; completion via TENSORcnt.
// ---------------------------------------------------------------------------
__device__ __forceinline__ void tdm_load_2d(unsigned int lds_addr,
                                            const void* gsrc,
                                            unsigned int tile_d0,
                                            unsigned int tile_d1,
                                            unsigned int stride_elems) {
  unsigned long long ga = (unsigned long long)(uintptr_t)gsrc;
  u32x4 g0;
  g0[0] = 1u;                                   // count=1 (valid user D#)
  g0[1] = lds_addr;                             // LDS byte address
  g0[2] = (unsigned int)ga;                     // global_addr[31:0]
  g0[3] = (unsigned int)(ga >> 32) | (2u << 30);// global_addr[56:32] | type=2
  i32x8 g1;
  g1[0] = 1 << 16;                              // data_size=1 -> 2-byte elems
  g1[1] = (int)0xFFFF0000u;                     // tensor_dim0[15:0]=0xFFFF
  g1[2] = (int)0xFFFF7FFFu;                     // tensor_dim0 hi / dim1 lo
  g1[3] = (int)(0x7FFFu | (tile_d0 << 16));     // tensor_dim1 hi | tile_dim0
  g1[4] = (int)(tile_d1 & 0xFFFFu);             // tile_dim1 (tile_dim2 = 0)
  g1[5] = (int)stride_elems;                    // tensor_dim0_stride[31:0]
  g1[6] = 0;                                    // stride hi / dim1_stride lo
  g1[7] = 0;
  i32x4 z4 = {0, 0, 0, 0};
#if __clang_major__ >= 23
  i32x8 z8 = {0, 0, 0, 0, 0, 0, 0, 0};
  __builtin_amdgcn_tensor_load_to_lds(g0, g1, z4, z4, z8, 0);
#else
  __builtin_amdgcn_tensor_load_to_lds(g0, g1, z4, z4, 0);
#endif
}

// ---------------------------------------------------------------------------
// Embedding: x[b,t,:] = sum_s rvq[s, tok[b,t,s], :] + pos_embed[pos[b,t], :]
//                       + intent[b, :]
// ---------------------------------------------------------------------------
__global__ void embed_kernel(const int* __restrict__ tok,
                             const int* __restrict__ pos,
                             const float* __restrict__ intent,
                             const float* __restrict__ rvq,
                             const float* __restrict__ pe,
                             float* __restrict__ x) {
  int row = blockIdx.x;            // 0..kM-1
  int b   = row / kT;
  int p   = pos[row];
  for (int d = threadIdx.x; d < kD; d += blockDim.x) {
    float acc = pe[(size_t)p * kD + d] + intent[(size_t)b * kD + d];
#pragma unroll
    for (int s = 0; s < kS; ++s) {
      int t = tok[(size_t)row * kS + s];
      acc += rvq[((size_t)s * kV + t) * kD + d];
    }
    x[(size_t)row * kD + d] = acc;
  }
}

// ---------------------------------------------------------------------------
// RMSNorm over D=768: one block (256 threads) per row, bf16 output for WMMA A.
// ---------------------------------------------------------------------------
__global__ __launch_bounds__(256)
void rmsnorm_kernel(const float* __restrict__ x, const float* __restrict__ w,
                    __bf16* __restrict__ out) {
  __shared__ float sred[8];
  int row = blockIdx.x;
  const float* xr = x + (size_t)row * kD;
  int t = threadIdx.x;
  float v0 = xr[t], v1 = xr[t + 256], v2 = xr[t + 512];
  float ss = v0 * v0 + v1 * v1 + v2 * v2;
#pragma unroll
  for (int o = 16; o >= 1; o >>= 1) ss += __shfl_xor(ss, o, 32);
  if ((t & 31) == 0) sred[t >> 5] = ss;
  __syncthreads();
  float tot = 0.f;
#pragma unroll
  for (int i = 0; i < 8; ++i) tot += sred[i];
  float rstd = rsqrtf(tot / (float)kD + 1e-6f);
  __bf16* orow = out + (size_t)row * kD;
  orow[t]       = (__bf16)(v0 * rstd * w[t]);
  orow[t + 256] = (__bf16)(v1 * rstd * w[t + 256]);
  orow[t + 512] = (__bf16)(v2 * rstd * w[t + 512]);
}

// ---------------------------------------------------------------------------
// Tiled WMMA GEMM:  C[M,N] (+resid) = A_bf16[M,K] * W_fp32[N,K]^T
//   block: 256 thr = 8 waves, tile 128(M) x 128(N), K-step 64, double-buffered
//   wave:  32x64 output = 2x4 wmma accum; 16 WMMAs per staged tile
//   A tile: TDM DMA (global bf16 -> LDS), issued by wave 0, next-tile prefetch
//   W tile: cooperative fp32->bf16 VALU convert, overlapped with compute+DMA
// MODE 0: Cf = AB     MODE 1: Cf = resid + AB
// MODE 3: Cbf = silu(resid) * AB     (bf16 out, SwiGLU fuse)
// ---------------------------------------------------------------------------
template <int MODE>
__global__ __launch_bounds__(256)
void gemm_kernel(const __bf16* __restrict__ A, const float* __restrict__ W,
                 const float* __restrict__ resid, float* __restrict__ Cf,
                 __bf16* __restrict__ Cbf, int N, int K) {
  __shared__ __attribute__((aligned(16))) __bf16 sA[2][128 * 64];
  __shared__ __attribute__((aligned(16))) __bf16 sB[2][128 * 64];
  int tid   = threadIdx.x;
  int lane  = tid & 31, wave = tid >> 5;
  int wm    = wave & 3, wn = wave >> 2;       // 4 x 2 wave grid
  int laneN = lane & 15, hiHalf = lane >> 4;
  int base0 = hiHalf ? 8 : 0;
  int m0 = blockIdx.y * 128;
  int n0 = blockIdx.x * 128;

  const f32x8 zero = {0.f, 0.f, 0.f, 0.f, 0.f, 0.f, 0.f, 0.f};
  f32x8 acc[2][4];
#pragma unroll
  for (int i = 0; i < 2; ++i)
#pragma unroll
    for (int j = 0; j < 4; ++j) acc[i][j] = zero;

  int wrow = tid >> 1, wseg = tid & 1;   // W tile: 128 rows x 2 segs of 32 f32
  auto stageW = [&](int k0, __bf16* dstbuf) {
    const float* wsrc = W + (size_t)(n0 + wrow) * K + k0 + wseg * 32;
    __bf16* dst = dstbuf + wrow * 64 + wseg * 32;
#pragma unroll
    for (int q = 0; q < 8; ++q) {
      float4 f = *(const float4*)(wsrc + q * 4);
      dst[q * 4 + 0] = (__bf16)f.x;
      dst[q * 4 + 1] = (__bf16)f.y;
      dst[q * 4 + 2] = (__bf16)f.z;
      dst[q * 4 + 3] = (__bf16)f.w;
    }
  };

  // prologue: stage K-tile 0 into buffer 0
  if (tid < 32)
    tdm_load_2d((unsigned int)(uintptr_t)&sA[0][0], A + (size_t)m0 * K,
                64u, 128u, (unsigned int)K);
  stageW(0, &sB[0][0]);
  if (tid < 32) __builtin_amdgcn_s_wait_tensorcnt(0);
  __syncthreads();

  int nsteps = K >> 6;
  for (int step = 0; step < nsteps; ++step) {
    int cur = step & 1, nxt = cur ^ 1;
    if (step + 1 < nsteps) {           // prefetch next tile into the alt buffer
      int k1 = (step + 1) << 6;
      if (tid < 32)
        tdm_load_2d((unsigned int)(uintptr_t)&sA[nxt][0],
                    A + (size_t)m0 * K + k1, 64u, 128u, (unsigned int)K);
      stageW(k1, &sB[nxt][0]);
    }
    const __bf16* cA = &sA[cur][0];
    const __bf16* cB = &sB[cur][0];
#pragma unroll
    for (int s = 0; s < 2; ++s) {      // two K-substeps of 32
      bf16x16 aF[2], bF[4];
#pragma unroll
      for (int i = 0; i < 2; ++i) {
        const __bf16* p = cA + (wm * 32 + i * 16 + laneN) * 64 + s * 32 + base0;
        aF[i] = ldfrag(p, p + 16);
      }
#pragma unroll
      for (int j = 0; j < 4; ++j) {
        const __bf16* p = cB + (wn * 64 + j * 16 + laneN) * 64 + s * 32 + base0;
        bF[j] = ldfrag(p, p + 16);
      }
#pragma unroll
      for (int i = 0; i < 2; ++i)
#pragma unroll
        for (int j = 0; j < 4; ++j)
          acc[i][j] = wmma_bf16(aF[i], bF[j], acc[i][j]);
    }
    if (tid < 32 && step + 1 < nsteps) __builtin_amdgcn_s_wait_tensorcnt(0);
    __syncthreads();
  }

  // epilogue: C layout — vgpr r holds rows r (lanes 0-15) / r+8 (lanes 16-31)
#pragma unroll
  for (int i = 0; i < 2; ++i)
#pragma unroll
    for (int j = 0; j < 4; ++j) {
      int col = n0 + wn * 64 + j * 16 + laneN;
#pragma unroll
      for (int r = 0; r < 8; ++r) {
        int row = m0 + wm * 32 + i * 16 + r + 8 * hiHalf;
        size_t idx = (size_t)row * N + col;
        float v = acc[i][j][r];
        if (MODE == 0) {
          Cf[idx] = v;
        } else if (MODE == 1) {
          Cf[idx] = resid[idx] + v;
        } else {  // MODE 3: SwiGLU fuse
          float g  = resid[idx];
          float sg = g / (1.f + __expf(-g));
          Cbf[idx] = (__bf16)(sg * v);
        }
      }
    }
}

// ---------------------------------------------------------------------------
// Split fused QKV (fp32 [M, 3*D]) into head-major bf16 q/k/v [B,H,T,64].
// ---------------------------------------------------------------------------
__global__ __launch_bounds__(256)
void qkv_split_kernel(const float* __restrict__ qkv, __bf16* __restrict__ qb,
                      __bf16* __restrict__ kb, __bf16* __restrict__ vb) {
  int row = blockIdx.x;
  int b = row / kT, t = row % kT;
  for (int e = threadIdx.x; e < 3 * kD; e += blockDim.x) {
    float val = qkv[(size_t)row * (3 * kD) + e];
    int sec = e / kD, rem = e % kD;
    int h = rem >> 6, dh = rem & 63;
    __bf16* dst = (sec == 0) ? qb : ((sec == 1) ? kb : vb);
    dst[(((size_t)b * kH + h) * kT + t) * kDH + dh] = (__bf16)val;
  }
}

// ---------------------------------------------------------------------------
// Flash attention: block = (qblk, h, b), 128 thr = 4 waves x 16 q-rows.
// KV tiles of 32, double-buffered: next K tile via TDM + next V transpose by
// the VALU while the matrix pipe computes the current tile.  Online softmax
// with DPP reductions; QK^T and P*V via WMMA.  Output bf16 into [M, D].
// ---------------------------------------------------------------------------
__global__ __launch_bounds__(128)
void attn_kernel(const __bf16* __restrict__ qb, const __bf16* __restrict__ kb,
                 const __bf16* __restrict__ vb, __bf16* __restrict__ obf) {
  __shared__ __attribute__((aligned(16))) __bf16 sK[2][32 * 64];   // [kv][dh]
  __shared__ __attribute__((aligned(16))) __bf16 sVt[2][64 * 32];  // [dh][kv]
  __shared__ __attribute__((aligned(16))) __bf16 sP[4 * 16 * 32];  // per wave
  const float scale = 0.125f;  // DH^-0.5

  int tid  = threadIdx.x;
  int wave = tid >> 5, lane = tid & 31;
  int laneN = lane & 15, hiHalf = lane >> 4;
  int base0 = hiHalf ? 8 : 0;
  int b = blockIdx.z, h = blockIdx.y, qblk = blockIdx.x;
  int q0 = qblk * 64 + wave * 16;
  size_t headBase = ((size_t)b * kH + h) * kT;

  // Q fragments for both DH 32-chunks, held for the whole KV loop
  const __bf16* qrow = qb + (headBase + q0 + laneN) * kDH;
  bf16x16 qf0 = ldfrag(qrow + base0,      qrow + base0 + 16);
  bf16x16 qf1 = ldfrag(qrow + 32 + base0, qrow + 32 + base0 + 16);

  const f32x8 zero = {0.f, 0.f, 0.f, 0.f, 0.f, 0.f, 0.f, 0.f};
  f32x8 oacc[4];
  float rmax[8], rsum[8];
#pragma unroll
  for (int nt = 0; nt < 4; ++nt) oacc[nt] = zero;
#pragma unroll
  for (int r = 0; r < 8; ++r) { rmax[r] = -3.0e38f; rsum[r] = 0.f; }

  int ldrow = tid >> 2, ldseg = tid & 3;
  auto stageV = [&](int kv0, __bf16* dstbuf) {   // V tile transposed [dh][kv]
    const __bf16* vsrc = vb + (headBase + kv0 + ldrow) * kDH + ldseg * 16;
#pragma unroll
    for (int e = 0; e < 16; ++e)
      dstbuf[(ldseg * 16 + e) * 32 + ldrow] = vsrc[e];
  };

  int ntiles = 2 * (qblk + 1);   // causal: only kv <= q-block end

  // prologue: stage KV tile 0
  if (tid < 32)
    tdm_load_2d((unsigned int)(uintptr_t)&sK[0][0], kb + headBase * kDH,
                64u, 32u, 64u);
  stageV(0, &sVt[0][0]);
  if (tid < 32) __builtin_amdgcn_s_wait_tensorcnt(0);
  __syncthreads();

  for (int kt = 0; kt < ntiles; ++kt) {
    int kv0 = kt * 32;
    int cur = kt & 1, nxt = cur ^ 1;
    if (kt + 1 < ntiles) {             // prefetch next KV tile
      if (tid < 32)
        tdm_load_2d((unsigned int)(uintptr_t)&sK[nxt][0],
                    kb + (headBase + kv0 + 32) * kDH, 64u, 32u, 64u);
      stageV(kv0 + 32, &sVt[nxt][0]);
    }
    const __bf16* cK = &sK[cur][0];
    const __bf16* cV = &sVt[cur][0];

    // S = Q * K^T : batch all fragment loads, then the WMMA burst
    const __bf16* p0 = cK + laneN * 64 + base0;
    const __bf16* p1 = cK + (16 + laneN) * 64 + base0;
    bf16x16 b0 = ldfrag(p0,      p0 + 16);
    bf16x16 b1 = ldfrag(p1,      p1 + 16);
    bf16x16 c0 = ldfrag(p0 + 32, p0 + 48);
    bf16x16 c1 = ldfrag(p1 + 32, p1 + 48);
    f32x8 s0 = zero, s1 = zero;
    s0 = wmma_bf16(qf0, b0, s0);
    s1 = wmma_bf16(qf0, b1, s1);
    s0 = wmma_bf16(qf1, c0, s0);
    s1 = wmma_bf16(qf1, c1, s1);

    // causal mask + online softmax (row stats live in the row's lane-half)
    __bf16* pw = &sP[wave * 16 * 32];
    float pscale[8];
#pragma unroll
    for (int r = 0; r < 8; ++r) {
      int qg = q0 + r + 8 * hiHalf;
      float v0 = s0[r] * scale; if (kv0 + laneN > qg)      v0 = -3.0e38f;
      float v1 = s1[r] * scale; if (kv0 + 16 + laneN > qg) v1 = -3.0e38f;
      float mx = red16_max(fmaxf(v0, v1));
      float mnew = fmaxf(rmax[r], mx);
      float sc   = __expf(rmax[r] - mnew);
      pscale[r]  = sc;
      rmax[r]    = mnew;
      float p0e = __expf(v0 - mnew);
      float p1e = __expf(v1 - mnew);
      rsum[r] = rsum[r] * sc + red16_sum(p0e + p1e);
      int lr = r + 8 * hiHalf;
      pw[lr * 32 + laneN]      = (__bf16)p0e;   // C-layout -> LDS
      pw[lr * 32 + 16 + laneN] = (__bf16)p1e;
    }
#pragma unroll
    for (int nt = 0; nt < 4; ++nt)
#pragma unroll
      for (int r = 0; r < 8; ++r) oacc[nt][r] *= pscale[r];

    // O += P * V  (A = P reloaded in A-layout; B = transposed V tile)
    const __bf16* pa = &pw[laneN * 32 + base0];
    bf16x16 aP = ldfrag(pa, pa + 16);
    bf16x16 bV[4];
#pragma unroll
    for (int nt = 0; nt < 4; ++nt) {
      const __bf16* pv = cV + (nt * 16 + laneN) * 32 + base0;
      bV[nt] = ldfrag(pv, pv + 16);
    }
#pragma unroll
    for (int nt = 0; nt < 4; ++nt) oacc[nt] = wmma_bf16(aP, bV[nt], oacc[nt]);

    if (tid < 32 && kt + 1 < ntiles) __builtin_amdgcn_s_wait_tensorcnt(0);
    __syncthreads();
  }

#pragma unroll
  for (int r = 0; r < 8; ++r) rsum[r] = 1.f / rsum[r];
#pragma unroll
  for (int nt = 0; nt < 4; ++nt) {
    int col = h * kDH + nt * 16 + laneN;
#pragma unroll
    for (int r = 0; r < 8; ++r) {
      int tokIdx = q0 + r + 8 * hiHalf;
      obf[((size_t)(b * kT + tokIdx)) * kD + col] =
          (__bf16)(oacc[nt][r] * rsum[r]);
    }
  }
}

// ---------------------------------------------------------------------------
// Final RMSNorm(last token) + per-stage logits: logits[s,b,v] = ln(b) . hw[s,v]
// 64 blocks: each handles 256 vocab entries for one (s,b).
// ---------------------------------------------------------------------------
__global__ __launch_bounds__(256)
void head_kernel(const float* __restrict__ x, const float* __restrict__ normf,
                 const float* __restrict__ hw, float* __restrict__ out) {
  __shared__ float ln[kD];
  __shared__ float sred[8];
  int blk = blockIdx.x;        // = ((s*2)+b)*4 + vc
  int s  = blk >> 3;
  int b  = (blk >> 2) & 1;
  int vc = blk & 3;
  int v  = vc * 256 + threadIdx.x;
  int t  = threadIdx.x;

  const float* xr = x + ((size_t)(b * kT + kT - 1)) * kD;
  float v0 = xr[t], v1 = xr[t + 256], v2 = xr[t + 512];
  float ss = v0 * v0 + v1 * v1 + v2 * v2;
#pragma unroll
  for (int o = 16; o >= 1; o >>= 1) ss += __shfl_xor(ss, o, 32);
  if ((t & 31) == 0) sred[t >> 5] = ss;
  __syncthreads();
  float tot = 0.f;
#pragma unroll
  for (int i = 0; i < 8; ++i) tot += sred[i];
  float rstd = rsqrtf(tot / (float)kD + 1e-6f);
  ln[t]       = v0 * rstd * normf[t];
  ln[t + 256] = v1 * rstd * normf[t + 256];
  ln[t + 512] = v2 * rstd * normf[t + 512];
  __syncthreads();

  const float* hr = hw + ((size_t)s * kV + v) * kD;
  float acc = 0.f;
  for (int d = 0; d < kD; ++d) acc += ln[d] * hr[d];
  out[((size_t)s * kB + b) * kV + v] = acc;
}

// ---------------------------------------------------------------------------
// Host-side orchestration (graph-capture safe: only kernel launches).
// ---------------------------------------------------------------------------
extern "C" void kernel_launch(void* const* d_in, const int* in_sizes, int n_in,
                              void* d_out, int out_size, void* d_ws,
                              size_t ws_size, hipStream_t stream) {
  (void)in_sizes; (void)n_in; (void)out_size; (void)ws_size;
  const int*   tok    = (const int*)  d_in[0];
  const int*   pos    = (const int*)  d_in[1];
  const float* intent = (const float*)d_in[2];
  const float* rvq    = (const float*)d_in[3];
  const float* pe     = (const float*)d_in[4];
  const float* n1w    = (const float*)d_in[5];
  const float* n2w    = (const float*)d_in[6];
  const float* qkvw   = (const float*)d_in[7];
  const float* projw  = (const float*)d_in[8];
  const float* gatew  = (const float*)d_in[9];
  const float* upw    = (const float*)d_in[10];
  const float* downw  = (const float*)d_in[11];
  const float* normf  = (const float*)d_in[12];
  const float* headsw = (const float*)d_in[13];
  float* out = (float*)d_out;

  // workspace carve-up (256B aligned; ~80 MB total)
  char* ws = (char*)d_ws;
  size_t off = 0;
  auto carve = [&](size_t bytes) -> char* {
    char* p = ws + off;
    off += (bytes + 255) & ~(size_t)255;
    return p;
  };
  float*  x    = (float*) carve((size_t)kM * kD * 4);
  float*  qkv  = (float*) carve((size_t)kM * 3 * kD * 4);
  float*  gbuf = (float*) carve((size_t)kM * kFF * 4);
  __bf16* hbf  = (__bf16*)carve((size_t)kM * kD * 2);
  __bf16* obf  = (__bf16*)carve((size_t)kM * kD * 2);
  __bf16* pbuf = (__bf16*)carve((size_t)kM * kFF * 2);
  __bf16* qb   = (__bf16*)carve((size_t)kB * kH * kT * kDH * 2);
  __bf16* kbuf = (__bf16*)carve((size_t)kB * kH * kT * kDH * 2);
  __bf16* vbuf = (__bf16*)carve((size_t)kB * kH * kT * kDH * 2);

  embed_kernel<<<kM, 256, 0, stream>>>(tok, pos, intent, rvq, pe, x);

  for (int l = 0; l < kL; ++l) {
    rmsnorm_kernel<<<kM, 256, 0, stream>>>(x, n1w + (size_t)l * kD, hbf);
    // QKV: [M,768] x [2304,768]^T
    gemm_kernel<0><<<dim3(18, 16), 256, 0, stream>>>(
        hbf, qkvw + (size_t)l * 3 * kD * kD, nullptr, qkv, nullptr,
        3 * kD, kD);
    qkv_split_kernel<<<kM, 256, 0, stream>>>(qkv, qb, kbuf, vbuf);
    attn_kernel<<<dim3(kT / 64, kH, kB), 128, 0, stream>>>(qb, kbuf, vbuf, obf);
    // x += O * proj^T
    gemm_kernel<1><<<dim3(6, 16), 256, 0, stream>>>(
        obf, projw + (size_t)l * kD * kD, x, x, nullptr, kD, kD);
    rmsnorm_kernel<<<kM, 256, 0, stream>>>(x, n2w + (size_t)l * kD, hbf);
    // gate (plain fp32), then up fused with silu(gate)*up -> bf16
    gemm_kernel<0><<<dim3(24, 16), 256, 0, stream>>>(
        hbf, gatew + (size_t)l * kFF * kD, nullptr, gbuf, nullptr,
        kFF, kD);
    gemm_kernel<3><<<dim3(24, 16), 256, 0, stream>>>(
        hbf, upw + (size_t)l * kFF * kD, gbuf, nullptr, pbuf,
        kFF, kD);
    // x += (silu(g)*u) * down^T
    gemm_kernel<1><<<dim3(6, 16), 256, 0, stream>>>(
        pbuf, downw + (size_t)l * kD * kFF, x, x, nullptr, kD, kFF);
  }

  head_kernel<<<kS * kB * (kV / 256), 256, 0, stream>>>(x, normf, headsw, out);
}
